// ParagraphGATInference_23965917512221
// MI455X (gfx1250) — matrix operand order repaired and checked by reading, not compiled, not run on hardware
//
#include <hip/hip_runtime.h>

typedef __attribute__((ext_vector_type(16))) _Float16 v16h;
typedef __attribute__((ext_vector_type(8)))  float    v8f;
typedef __attribute__((ext_vector_type(4)))  _Float16 h4;

#define DIM   128
#define HEADS 4
#define HD    512        // HEADS * DIM
#define NEG_SLOPE 0.2f
#define LN_EPS    1e-5f

// ---------------------------------------------------------------- utilities

__device__ __forceinline__ float waveSum32(float v) {
#pragma unroll
  for (int off = 16; off > 0; off >>= 1) v += __shfl_xor(v, off, 32);
  return v;
}

// float atomic max via sign-punning (valid when location initialized to a
// negative float such as -1e30f).
__device__ __forceinline__ void atomicMaxF(float* addr, float val) {
  if (val >= 0.f) atomicMax((int*)addr, __float_as_int(val));
  else            atomicMin((unsigned int*)addr, (unsigned int)__float_as_int(val));
}

__global__ void k_fill(float* __restrict__ p, float v, int n) {
  int i = blockIdx.x * blockDim.x + threadIdx.x;
  if (i < n) p[i] = v;
}

__global__ void k_f32_to_f16(const float* __restrict__ in, _Float16* __restrict__ out, int n) {
  int i = blockIdx.x * blockDim.x + threadIdx.x;
  if (i < n) out[i] = (_Float16)in[i];
}

// ---------------------------------------------------------------- W packing
// Pack W (f32, 128 x 512 row-major, K x N) into WMMA B-fragment order:
// frag f = nTile*4 + kk covers cols nTile*16..+15, K kk*32..+31.
// Within a frag: lane L, element i  ->  B[kk*32 + (L>=16?8:0) + (i&7) + (i>=8?16:0)][nTile*16 + (L&15)]
// stored contiguously: Bp[f*512 + L*16 + i].
// A block's 16 fragments (4 nTiles x 4 kk) are contiguous: Bp + by*16*512.
__global__ void k_packB(const float* __restrict__ W, _Float16* __restrict__ Bp) {
  int idx = blockIdx.x * blockDim.x + threadIdx.x;   // 0 .. 65535
  if (idx >= DIM * HD) return;
  int i     =  idx        & 15;
  int lane  = (idx >> 4)  & 31;
  int kk    = (idx >> 9)  & 3;
  int nTile =  idx >> 11;                            // 0..31
  int k = kk * 32 + ((lane >> 4) << 3) + (i & 7) + ((i >> 3) << 4);
  int n = nTile * 16 + (lane & 15);
  Bp[idx] = (_Float16)W[k * HD + n];
}

// ---------------------------------------------------------------- WMMA GEMM
// C16[nrows x 512] = A16[nrows x 128] @ B16[128 x 512], f32 accumulate,
// f16 output (feeds only softmax-attention gathers -> precision is ample).
// block = 256 threads = 8 waves; wave w -> rows m0 = bx*128 + w*16,
// cols blockIdx.y*64 .. +63 (4 x 16x16 tiles), K loop of 4 x 32.
// B fragments staged once per block into LDS (16 KB) to kill the 8x
// per-wave reload of identical weight data and decouple WMMA from VMEM.
__global__ __launch_bounds__(256)
void k_gemm_wmma(const _Float16* __restrict__ A, const _Float16* __restrict__ Bp,
                 _Float16* __restrict__ C, int nrows) {
  __shared__ _Float16 sB[16 * 512];                       // 16 KB
  const _Float16* gB = Bp + (size_t)blockIdx.y * (16 * 512);
#pragma unroll
  for (int i = 0; i < 4; ++i) {
    int off = (threadIdx.x + i * 256) * 8;                // 8 halves per b128
    *(float4*)(sB + off) = *(const float4*)(gB + off);
  }
  __syncthreads();

  const int wave = threadIdx.x >> 5;
  const int lane = threadIdx.x & 31;
  const int m0   = blockIdx.x * 128 + wave * 16;
  if (m0 >= nrows) return;                                // wave-uniform, after barrier

  // A fragment: lanes 0-15 row m0+lane, K chunks {0..7,16..23};
  //             lanes 16-31 row m0+lane-16, K chunks {8..15,24..31}
  int m = m0 + (lane & 15);
  if (m >= nrows) m = nrows - 1;                          // clamp (stores guarded)
  const int kbA = (lane >> 4) << 3;                       // 0 or 8 halves
  const _Float16* arow = A + (size_t)m * DIM;

  v8f acc[4] = {};
#pragma unroll
  for (int kk = 0; kk < 4; ++kk) {
    v16h a;
    ((float4*)&a)[0] = *(const float4*)(arow + kk * 32 + kbA);
    ((float4*)&a)[1] = *(const float4*)(arow + kk * 32 + kbA + 16);
#pragma unroll
    for (int t = 0; t < 4; ++t) {
      const _Float16* bfrag = sB + (((t << 2) + kk) << 9) + lane * 16;
      v16h b;
      ((float4*)&b)[0] = ((const float4*)bfrag)[0];
      ((float4*)&b)[1] = ((const float4*)bfrag)[1];
      acc[t] = __builtin_amdgcn_wmma_f32_16x16x32_f16(
          false, a, false, b, (short)0, acc[t], false, false);
    }
  }
  // C layout: VGPR r -> lanes 0-15: row m0+r, lanes 16-31: row m0+8+r; N = lane&15
  const int rowb = m0 + ((lane >> 4) << 3);
  const int col  = blockIdx.y * 64 + (lane & 15);
#pragma unroll
  for (int t = 0; t < 4; ++t)
#pragma unroll
    for (int r = 0; r < 8; ++r)
      if (rowb + r < nrows)
        C[(size_t)(rowb + r) * HD + col + t * 16] = (_Float16)acc[t][r];
}

// ---------------------------------------------------------------- edge phase
// One wave per edge: e_score[e][h] = sum_d att[h][d]*leakyrelu(xl[s][h][d]+xr[d][h][d])
// xl/xr gathered as f16 (halves the L2 gather traffic), math in f32.
__global__ __launch_bounds__(256)
void k_edge_scores(const _Float16* __restrict__ xl, const _Float16* __restrict__ xr,
                   const int* __restrict__ src, const int* __restrict__ dst,
                   const float* __restrict__ att, float* __restrict__ score,
                   float* __restrict__ emax, int E, int EE) {
  const int e = blockIdx.x * 8 + (threadIdx.x >> 5);
  if (e >= EE) return;
  const int lane = threadIdx.x & 31;
  const int s = (e < E) ? src[e] : (e - E);
  const int d = (e < E) ? dst[e] : (e - E);
  const _Float16* pl = xl + (size_t)s * HD;
  const _Float16* pr = xr + (size_t)d * HD;
  float sums[HEADS];
#pragma unroll
  for (int h = 0; h < HEADS; ++h) {
    float acc = 0.f;
#pragma unroll
    for (int j = 0; j < 4; ++j) {
      const int dd = lane + j * 32;
      float v = (float)pl[h * DIM + dd] + (float)pr[h * DIM + dd];
      v = (v > 0.f) ? v : v * NEG_SLOPE;
      acc += v * att[h * DIM + dd];
    }
    sums[h] = waveSum32(acc);
  }
  if (lane == 0) {
#pragma unroll
    for (int h = 0; h < HEADS; ++h) {
      score[(size_t)e * HEADS + h] = sums[h];
      atomicMaxF(&emax[d * HEADS + h], sums[h]);
    }
  }
}

// One thread per (edge, head): ex = exp(score - emax[dst]); denom += ex (in place)
__global__ void k_edge_exp(const int* __restrict__ dst, float* __restrict__ score,
                           const float* __restrict__ emax, float* __restrict__ denom,
                           int E, int EE) {
  int idx = blockIdx.x * blockDim.x + threadIdx.x;
  if (idx >= EE * HEADS) return;
  int e = idx >> 2, h = idx & 3;
  int d = (e < E) ? dst[e] : (e - E);
  float ex = __expf(score[idx] - emax[d * HEADS + h]);
  score[idx] = ex;
  atomicAdd(&denom[d * HEADS + h], ex);
}

// One wave per edge: agg[dst][d] += (1/H) * sum_h alpha_h * xl[src][h][d]
// (head mean folded in -> one atomicAdd per output float; xl gathered f16)
__global__ __launch_bounds__(256)
void k_scatter(const _Float16* __restrict__ xl, const int* __restrict__ src,
               const int* __restrict__ dst, const float* __restrict__ ex,
               const float* __restrict__ denom, float* __restrict__ agg,
               int E, int EE) {
  const int e = blockIdx.x * 8 + (threadIdx.x >> 5);
  if (e >= EE) return;
  const int lane = threadIdx.x & 31;
  const int s = (e < E) ? src[e] : (e - E);
  const int d = (e < E) ? dst[e] : (e - E);
  float alpha[HEADS];
#pragma unroll
  for (int h = 0; h < HEADS; ++h)
    alpha[h] = 0.25f * ex[(size_t)e * HEADS + h] / (denom[d * HEADS + h] + 1e-16f);
  const _Float16* pl = xl + (size_t)s * HD + lane * 4;
  float rx = 0.f, ry = 0.f, rz = 0.f, rw = 0.f;
#pragma unroll
  for (int h = 0; h < HEADS; ++h) {
    h4 v = *(const h4*)(pl + h * DIM);                    // 8-byte gather
    rx += alpha[h] * (float)v[0]; ry += alpha[h] * (float)v[1];
    rz += alpha[h] * (float)v[2]; rw += alpha[h] * (float)v[3];
  }
  float* po = agg + (size_t)d * DIM + lane * 4;
  atomicAdd(po + 0, rx); atomicAdd(po + 1, ry);
  atomicAdd(po + 2, rz); atomicAdd(po + 3, rw);
}

// ---------------------------------------------------------------- finalize
// One wave per node: y = LN(agg + bias)*g + be + resid ; optional ReLU;
// writes f32 output and (optionally) f16 activations for the next GEMM.
__global__ __launch_bounds__(256)
void k_finalize(const float* __restrict__ agg, const float* __restrict__ bias,
                const float* __restrict__ g, const float* __restrict__ be,
                const float* __restrict__ resid, float* __restrict__ outF,
                _Float16* __restrict__ out16, int applyRelu, int nrows) {
  const int node = blockIdx.x * 8 + (threadIdx.x >> 5);
  if (node >= nrows) return;
  const int lane = threadIdx.x & 31;
  float4 v  = *(const float4*)(agg + (size_t)node * DIM + lane * 4);
  float4 bb = *(const float4*)(bias + lane * 4);
  v.x += bb.x; v.y += bb.y; v.z += bb.z; v.w += bb.w;
  const float mu  = waveSum32(v.x + v.y + v.z + v.w) * (1.f / DIM);
  const float dx = v.x - mu, dy = v.y - mu, dz = v.z - mu, dw = v.w - mu;
  const float var = waveSum32(dx * dx + dy * dy + dz * dz + dw * dw) * (1.f / DIM);
  const float inv = rsqrtf(var + LN_EPS);
  float4 gg = *(const float4*)(g + lane * 4);
  float4 b2 = *(const float4*)(be + lane * 4);
  float4 rr = *(const float4*)(resid + (size_t)node * DIM + lane * 4);
  float4 y;
  y.x = dx * inv * gg.x + b2.x + rr.x;
  y.y = dy * inv * gg.y + b2.y + rr.y;
  y.z = dz * inv * gg.z + b2.z + rr.z;
  y.w = dw * inv * gg.w + b2.w + rr.w;
  if (applyRelu) {
    y.x = fmaxf(y.x, 0.f); y.y = fmaxf(y.y, 0.f);
    y.z = fmaxf(y.z, 0.f); y.w = fmaxf(y.w, 0.f);
  }
  *(float4*)(outF + (size_t)node * DIM + lane * 4) = y;
  if (out16) {
    _Float16* o = out16 + (size_t)node * DIM + lane * 4;
    o[0] = (_Float16)y.x; o[1] = (_Float16)y.y;
    o[2] = (_Float16)y.z; o[3] = (_Float16)y.w;
  }
}

// ---------------------------------------------------------------- launch
extern "C" void kernel_launch(void* const* d_in, const int* in_sizes, int n_in,
                              void* d_out, int out_size, void* d_ws, size_t ws_size,
                              hipStream_t stream) {
  const float* x    = (const float*)d_in[0];
  const int*   src  = (const int*)  d_in[1];
  const int*   dst  = (const int*)  d_in[2];
  const float* Wl1  = (const float*)d_in[3];
  const float* Wr1  = (const float*)d_in[4];
  const float* att1 = (const float*)d_in[5];
  const float* b1   = (const float*)d_in[6];
  const float* g1   = (const float*)d_in[7];
  const float* be1  = (const float*)d_in[8];
  const float* Wl2  = (const float*)d_in[9];
  const float* Wr2  = (const float*)d_in[10];
  const float* att2 = (const float*)d_in[11];
  const float* b2   = (const float*)d_in[12];
  const float* g2   = (const float*)d_in[13];
  const float* be2  = (const float*)d_in[14];

  const int N  = in_sizes[0] / DIM;
  const int E  = in_sizes[1];
  const int EE = E + N;

  char*  ws  = (char*)d_ws;
  size_t off = 0;
  auto alloc = [&](size_t bytes) -> void* {
    void* p = ws + off;
    off += (bytes + 255) & ~(size_t)255;
    return p;
  };
  _Float16* x16   = (_Float16*)alloc((size_t)N * DIM * 2);   // reused as h16
  _Float16* bp    = (_Float16*)alloc((size_t)4 * DIM * HD * 2);
  _Float16* xl    = (_Float16*)alloc((size_t)N * HD * 2);    // f16 projections
  _Float16* xr    = (_Float16*)alloc((size_t)N * HD * 2);
  float*    score = (float*)alloc((size_t)EE * HEADS * 4);
  float*    emax  = (float*)alloc((size_t)N * HEADS * 4);
  float*    denom = (float*)alloc((size_t)N * HEADS * 4);
  float*    agg   = (float*)alloc((size_t)N * DIM * 4);
  float*    h     = (float*)alloc((size_t)N * DIM * 4);

  _Float16* bpW[4] = {bp, bp + DIM * HD, bp + 2 * DIM * HD, bp + 3 * DIM * HD};
  const float* Wsrc[4] = {Wl1, Wr1, Wl2, Wr2};
  for (int i = 0; i < 4; ++i)
    k_packB<<<(DIM * HD + 255) / 256, 256, 0, stream>>>(Wsrc[i], bpW[i]);
  k_f32_to_f16<<<(N * DIM + 255) / 256, 256, 0, stream>>>(x, x16, N * DIM);

  const dim3 gemmGrid((N + 127) / 128, HD / 64);
  const int  edgeBlocks = (EE + 7) / 8;
  const int  nodeBlocks = (N + 7) / 8;

  for (int layer = 0; layer < 2; ++layer) {
    const _Float16* bl  = layer ? bpW[2] : bpW[0];
    const _Float16* br  = layer ? bpW[3] : bpW[1];
    const float* att    = layer ? att2 : att1;
    const float* bias   = layer ? b2   : b1;
    const float* gamma  = layer ? g2   : g1;
    const float* beta   = layer ? be2  : be1;
    const float* resid  = layer ? h    : x;
    float*       outF   = layer ? (float*)d_out : h;
    _Float16*    out16  = layer ? nullptr : x16;   // h16 for layer-2 GEMM
    const int    relu   = layer ? 0 : 1;

    k_gemm_wmma<<<gemmGrid, 256, 0, stream>>>(x16, bl, xl, N);
    k_gemm_wmma<<<gemmGrid, 256, 0, stream>>>(x16, br, xr, N);

    k_fill<<<(N * HEADS + 255) / 256, 256, 0, stream>>>(emax, -1e30f, N * HEADS);
    k_fill<<<(N * HEADS + 255) / 256, 256, 0, stream>>>(denom, 0.f, N * HEADS);
    k_fill<<<(N * DIM + 255) / 256, 256, 0, stream>>>(agg, 0.f, N * DIM);

    k_edge_scores<<<edgeBlocks, 256, 0, stream>>>(xl, xr, src, dst, att, score,
                                                  emax, E, EE);
    k_edge_exp<<<(EE * HEADS + 255) / 256, 256, 0, stream>>>(dst, score, emax,
                                                             denom, E, EE);
    k_scatter<<<edgeBlocks, 256, 0, stream>>>(xl, src, dst, score, denom, agg,
                                              E, EE);
    k_finalize<<<nodeBlocks, 256, 0, stream>>>(agg, bias, gamma, beta, resid,
                                               outF, out16, relu, N);
  }
}
